// AttnDecoder_30786325578156
// MI455X (gfx1250) — matrix operand order repaired and checked by reading, not compiled, not run on hardware
//
#include <hip/hip_runtime.h>
#include <hip/hip_bf16.h>

// ---------------------------------------------------------------------------
// Problem constants (match the reference)
// ---------------------------------------------------------------------------
#define VV   10000   // vocab
#define EE   256     // embed dim
#define HH   512     // hidden dim
#define DD   768     // encoder dim
#define BB   32      // batch
#define TT   128     // time steps
#define NN   196     // encoder positions

typedef __attribute__((ext_vector_type(8)))  _Float16 v8h;
typedef __attribute__((ext_vector_type(16))) _Float16 v16h;
typedef __attribute__((ext_vector_type(8)))  float    v8f;

// ---------------------------------------------------------------------------
// WMMA fragment loaders (wave32, v_wmma_f32_16x16x32_f16 layouts)
// A (16x32, row-major in memory): lane L -> row M=L&15,
//   elems 0..7  = K[khalf+0 .. khalf+7],  elems 8..15 = K[khalf+16 .. khalf+23]
//   khalf = (L>>4)*8    -> two b128 loads at p and p+16
// B via W^T (N-major, K contiguous): lane L -> col N=L&15,
//   elems 0..7 = K[koff+0 .. koff+7], elems 8..15 = K[koff+16 .. koff+23], koff=(L>>4)*8 -> two b128 loads
// ---------------------------------------------------------------------------
__device__ __forceinline__ v16h load_a_frag(const _Float16* p) {
    v8h lo = *reinterpret_cast<const v8h*>(p);
    v8h hi = *reinterpret_cast<const v8h*>(p + 16);
    return __builtin_shufflevector(lo, hi, 0,1,2,3,4,5,6,7,8,9,10,11,12,13,14,15);
}
__device__ __forceinline__ v16h load_b_frag(const _Float16* p) {
    v8h lo = *reinterpret_cast<const v8h*>(p);
    v8h hi = *reinterpret_cast<const v8h*>(p + 16);
    return __builtin_shufflevector(lo, hi, 0,1,2,3,4,5,6,7,8,9,10,11,12,13,14,15);
}

// ---------------------------------------------------------------------------
// Generic WMMA GEMM:  C[M][N] = act( A[M][K] * BT[N][K]^T + bias[N] )
// A row-major f16 (lda), BT row-major f16 (ldb), C f32 (ldc),
// optional f16 mirror C16 (ldc16). One 16x16 tile per wave, 4 waves/block.
// act: 0 = none, 1 = tanh
// ---------------------------------------------------------------------------
__global__ void k_gemm_wmma(const _Float16* __restrict__ A, int lda,
                            const _Float16* __restrict__ BT, int ldb,
                            const float* __restrict__ bias,
                            float* __restrict__ C, int ldc,
                            _Float16* __restrict__ C16, int ldc16,
                            int Mtiles, int Ntiles, int K, int act) {
    const int lane = threadIdx.x & 31;
    const int tile = blockIdx.x * 4 + (threadIdx.x >> 5);
    if (tile >= Mtiles * Ntiles) return;
    const int tm = tile % Mtiles;
    const int tn = tile / Mtiles;

    const int lrow  = lane & 15;
    const int khalfA = (lane >> 4) * 8;
    const int koffB  = (lane >> 4) * 8;

    const _Float16* Arow = A  + (size_t)(tm * 16 + lrow) * lda + khalfA;
    const _Float16* Brow = BT + (size_t)(tn * 16 + lrow) * ldb + koffB;

    v8f acc = {};
    for (int k0 = 0; k0 < K; k0 += 32) {
        __builtin_prefetch(Brow + k0 + 128, 0, 0);   // global_prefetch_b8 on weight stream
        v16h a = load_a_frag(Arow + k0);
        v16h b = load_b_frag(Brow + k0);
        acc = __builtin_amdgcn_wmma_f32_16x16x32_f16(
            /*neg_a=*/false, a, /*neg_b=*/false, b,
            /*c_mod=*/(short)0, acc, /*reuse_a=*/false, /*reuse_b=*/false);
    }

    const int n     = tn * 16 + lrow;
    const int mbase = tm * 16 + (lane >> 4) * 8;
    const float bs  = bias ? bias[n] : 0.0f;
#pragma unroll
    for (int r = 0; r < 8; ++r) {
        float vval = acc[r] + bs;
        if (act == 1) vval = tanhf(vval);
        C[(size_t)(mbase + r) * ldc + n] = vval;
        if (C16) C16[(size_t)(mbase + r) * ldc16 + n] = (_Float16)vval;
    }
}

// ---------------------------------------------------------------------------
// Setup kernels
// ---------------------------------------------------------------------------
// dst[n][dcol+k] = (f16) src[k][n]   (transpose + convert, src is K x N f32)
__global__ void k_transpose16(const float* __restrict__ src, _Float16* __restrict__ dst,
                              int K, int N, int dld, int dcol) {
    size_t idx = (size_t)blockIdx.x * blockDim.x + threadIdx.x;
    if (idx >= (size_t)K * N) return;
    int k = (int)(idx / N);
    int n = (int)(idx % N);
    dst[(size_t)n * dld + dcol + k] = (_Float16)src[(size_t)k * N + n];
}

__global__ void k_cvt16(const float* __restrict__ src, _Float16* __restrict__ dst, size_t n) {
    size_t idx = (size_t)blockIdx.x * blockDim.x + threadIdx.x;
    if (idx < n) dst[idx] = (_Float16)src[idx];
}

// emb16[b][t][e] = table[x[b][t]][e]
__global__ void k_emb_gather(const int* __restrict__ x, const float* __restrict__ table,
                             _Float16* __restrict__ emb16) {
    size_t idx = (size_t)blockIdx.x * blockDim.x + threadIdx.x;
    if (idx >= (size_t)BB * TT * EE) return;
    int tok = x[idx >> 8];            // EE == 256
    int e   = (int)(idx & 255);
    emb16[idx] = (_Float16)table[(size_t)tok * EE + e];
}

// pooled16[b][d] = mean_n enc[b][n][d]
__global__ void k_pooled(const float* __restrict__ enc, _Float16* __restrict__ pooled16) {
    int idx = blockIdx.x * blockDim.x + threadIdx.x;
    if (idx >= BB * DD) return;
    int b = idx / DD, d = idx % DD;
    float s = 0.f;
    const float* p = enc + (size_t)b * NN * DD + d;
    for (int n = 0; n < NN; ++n) s += p[(size_t)n * DD];
    pooled16[idx] = (_Float16)(s * (1.0f / NN));
}

__global__ void k_addbias(const float* __restrict__ a, const float* __restrict__ b,
                          float* __restrict__ o, int n) {
    int i = blockIdx.x * blockDim.x + threadIdx.x;
    if (i < n) o[i] = a[i] + b[i];
}

// ---------------------------------------------------------------------------
// Per-step kernels
// ---------------------------------------------------------------------------
// scores[b][n] = mask ? sum_h v[h]*tanh(hWh[b][h] + Ke[b*N+n][h]) : -1e9
__global__ void k_scores(const float* __restrict__ hWh, const float* __restrict__ Ke,
                         const float* __restrict__ vvec, const int* __restrict__ mask,
                         float* __restrict__ scores) {
    int wave = (blockIdx.x * blockDim.x + threadIdx.x) >> 5;
    int lane = threadIdx.x & 31;
    if (wave >= BB * NN) return;
    int b = wave / NN;
    const float* ke = Ke  + (size_t)wave * HH;
    const float* hw = hWh + (size_t)b * HH;
    float s = 0.f;
    for (int h = lane; h < HH; h += 32) s += vvec[h] * tanhf(hw[h] + ke[h]);
#pragma unroll
    for (int off = 16; off > 0; off >>= 1) s += __shfl_down(s, off, 32);
    if (lane == 0) scores[wave] = (mask[wave] == 0) ? -1.0e9f : s;
}

// softmax over N, context = alpha @ enc, assemble xcat = [emb_t | context | h] (f16)
__global__ void k_softmax_ctx(const float* __restrict__ scores, const float* __restrict__ enc,
                              const _Float16* __restrict__ emb16, const _Float16* __restrict__ h16,
                              _Float16* __restrict__ xcat, int t) {
    __shared__ float red[256];
    __shared__ float alpha[NN];
    const int b = blockIdx.x, tid = threadIdx.x;

    float val = (tid < NN) ? scores[b * NN + tid] : -3.0e38f;
    red[tid] = val; __syncthreads();
    for (int s = 128; s > 0; s >>= 1) {
        if (tid < s) red[tid] = fmaxf(red[tid], red[tid + s]);
        __syncthreads();
    }
    const float mx = red[0]; __syncthreads();
    float e = (tid < NN) ? __expf(val - mx) : 0.f;
    if (tid < NN) alpha[tid] = e;
    red[tid] = e; __syncthreads();
    for (int s = 128; s > 0; s >>= 1) {
        if (tid < s) red[tid] += red[tid + s];
        __syncthreads();
    }
    const float inv = 1.0f / red[0];
    if (tid < NN) alpha[tid] *= inv;
    __syncthreads();

    // context -> xcat[b][EE .. EE+DD)
    for (int d = tid; d < DD; d += 256) {
        float s = 0.f;
        const float* ep = enc + (size_t)b * NN * DD + d;
        for (int n = 0; n < NN; ++n) s += alpha[n] * ep[(size_t)n * DD];
        xcat[(size_t)b * 1536 + EE + d] = (_Float16)s;
    }
    // emb_t -> xcat[b][0 .. EE)
    if (tid < EE)
        xcat[(size_t)b * 1536 + tid] = emb16[((size_t)b * TT + t) * EE + tid];
    // h -> xcat[b][EE+DD .. 1536)
    for (int j = tid; j < HH; j += 256)
        xcat[(size_t)b * 1536 + EE + DD + j] = h16[(size_t)b * HH + j];
}

__device__ __forceinline__ float sigm(float x) { return 1.0f / (1.0f + __expf(-x)); }

// LSTM cell: gates[b][0..4H) in torch order (i,f,g,o)
__global__ void k_cell(const float* __restrict__ gates, float* __restrict__ h,
                       float* __restrict__ c, _Float16* __restrict__ h16) {
    int idx = blockIdx.x * blockDim.x + threadIdx.x;
    if (idx >= BB * HH) return;
    int b = idx >> 9, j = idx & 511;          // HH == 512
    const float* g = gates + (size_t)b * 4 * HH;
    float ig = g[j], fg = g[HH + j], gg = g[2 * HH + j], og = g[3 * HH + j];
    float cc = sigm(fg) * c[idx] + sigm(ig) * tanhf(gg);
    float hh = sigm(og) * tanhf(cc);
    c[idx] = cc; h[idx] = hh; h16[idx] = (_Float16)hh;
}

// ---------------------------------------------------------------------------
// Host-side orchestration
// ---------------------------------------------------------------------------
extern "C" void kernel_launch(void* const* d_in, const int* in_sizes, int n_in,
                              void* d_out, int out_size, void* d_ws, size_t ws_size,
                              hipStream_t stream) {
    (void)in_sizes; (void)n_in; (void)out_size; (void)ws_size;
    const int*   x        = (const int*)  d_in[0];
    const float* enc      = (const float*)d_in[1];
    const int*   mask     = (const int*)  d_in[2];
    const float* table    = (const float*)d_in[3];
    const float* We       = (const float*)d_in[4];
    const float* Wh       = (const float*)d_in[5];
    const float* vvec     = (const float*)d_in[6];
    const float* Wih      = (const float*)d_in[7];
    const float* Whh      = (const float*)d_in[8];
    const float* bih      = (const float*)d_in[9];
    const float* bhh      = (const float*)d_in[10];
    const float* fcW      = (const float*)d_in[11];
    const float* fcb      = (const float*)d_in[12];
    const float* W2h      = (const float*)d_in[13];
    const float* b2h      = (const float*)d_in[14];
    const float* W2c      = (const float*)d_in[15];
    const float* b2c      = (const float*)d_in[16];

    float* out    = (float*)d_out;
    float* hOut   = out + (size_t)BB * TT * VV;   // final h (B,H)
    float* cOut   = hOut + (size_t)BB * HH;       // final c (B,H)

    // workspace carve (256B aligned)
    char* w = (char*)d_ws;
    size_t off = 0;
    auto alloc = [&](size_t bytes) -> void* {
        void* p = w + off;
        off = (off + bytes + 255) & ~(size_t)255;
        return p;
    };
    _Float16* WhT    = (_Float16*)alloc((size_t)HH * HH * 2);          // [512][512]
    _Float16* WcombT = (_Float16*)alloc((size_t)(4*HH) * 1536 * 2);    // [2048][1536]
    _Float16* fcWT   = (_Float16*)alloc((size_t)VV * HH * 2);          // [10000][512]
    _Float16* WeT    = (_Float16*)alloc((size_t)HH * DD * 2);          // [512][768]
    _Float16* W2hT   = (_Float16*)alloc((size_t)HH * DD * 2);
    _Float16* W2cT   = (_Float16*)alloc((size_t)HH * DD * 2);
    _Float16* enc16  = (_Float16*)alloc((size_t)BB * NN * DD * 2);
    _Float16* emb16  = (_Float16*)alloc((size_t)BB * TT * EE * 2);
    _Float16* pool16 = (_Float16*)alloc((size_t)BB * DD * 2);
    float*    Ke     = (float*)   alloc((size_t)BB * NN * HH * 4);
    float*    hWh    = (float*)   alloc((size_t)BB * HH * 4);
    float*    scores = (float*)   alloc((size_t)BB * NN * 4);
    _Float16* xcat   = (_Float16*)alloc((size_t)BB * 1536 * 2);
    float*    gates  = (float*)   alloc((size_t)BB * 4 * HH * 4);
    _Float16* h16    = (_Float16*)alloc((size_t)BB * HH * 2);
    float*    gbias  = (float*)   alloc((size_t)4 * HH * 4);

    const dim3 blk(256);
    auto grid1 = [](size_t n) { return dim3((unsigned)((n + 255) / 256)); };
    auto gridT = [](int tiles) { return dim3((unsigned)((tiles + 3) / 4)); };
    const dim3 blkG(128);   // 4 waves per block for GEMM

    // ---- one-time setup -------------------------------------------------
    k_transpose16<<<grid1((size_t)HH*HH),      blk, 0, stream>>>(Wh,  WhT,    HH,   HH,     HH,   0);
    k_transpose16<<<grid1((size_t)1024*2048),  blk, 0, stream>>>(Wih, WcombT, 1024, 4*HH,   1536, 0);
    k_transpose16<<<grid1((size_t)HH*2048),    blk, 0, stream>>>(Whh, WcombT, HH,   4*HH,   1536, 1024);
    k_transpose16<<<grid1((size_t)HH*VV),      blk, 0, stream>>>(fcW, fcWT,   HH,   VV,     HH,   0);
    k_transpose16<<<grid1((size_t)DD*HH),      blk, 0, stream>>>(We,  WeT,    DD,   HH,     DD,   0);
    k_transpose16<<<grid1((size_t)DD*HH),      blk, 0, stream>>>(W2h, W2hT,   DD,   HH,     DD,   0);
    k_transpose16<<<grid1((size_t)DD*HH),      blk, 0, stream>>>(W2c, W2cT,   DD,   HH,     DD,   0);
    k_cvt16<<<grid1((size_t)BB*NN*DD), blk, 0, stream>>>(enc, enc16, (size_t)BB*NN*DD);
    k_emb_gather<<<grid1((size_t)BB*TT*EE), blk, 0, stream>>>(x, table, emb16);
    k_pooled<<<grid1((size_t)BB*DD), blk, 0, stream>>>(enc, pool16);
    k_addbias<<<grid1(4*HH), blk, 0, stream>>>(bih, bhh, gbias, 4*HH);

    // init: h0 = tanh(pooled@W2h + b2h), c0 = tanh(pooled@W2c + b2c)
    k_gemm_wmma<<<gridT(2*32), blkG, 0, stream>>>(pool16, DD, W2hT, DD, b2h,
                                                  hOut, HH, h16, HH, 2, 32, DD, 1);
    k_gemm_wmma<<<gridT(2*32), blkG, 0, stream>>>(pool16, DD, W2cT, DD, b2c,
                                                  cOut, HH, (_Float16*)nullptr, 0, 2, 32, DD, 1);
    // Ke = enc @ We  : (B*N=6272) x 768 x 512 -> 392 x 32 tiles
    k_gemm_wmma<<<gridT(392*32), blkG, 0, stream>>>(enc16, DD, WeT, DD, (const float*)nullptr,
                                                    Ke, HH, (_Float16*)nullptr, 0, 392, 32, DD, 0);

    // ---- sequential decode ---------------------------------------------
    for (int t = 0; t < TT; ++t) {
        // hWh = h @ Wh  (32x512x512)
        k_gemm_wmma<<<gridT(2*32), blkG, 0, stream>>>(h16, HH, WhT, HH, (const float*)nullptr,
                                                      hWh, HH, (_Float16*)nullptr, 0, 2, 32, HH, 0);
        // scores: one wave per (b,n)
        k_scores<<<dim3((BB*NN)/8), blk, 0, stream>>>(hWh, Ke, vvec, mask, scores);
        // softmax + context + xcat assembly
        k_softmax_ctx<<<dim3(BB), blk, 0, stream>>>(scores, enc, emb16, h16, xcat, t);
        // gates = xcat @ [Wih;Whh] + (bih+bhh)  (32x1536x2048)
        k_gemm_wmma<<<gridT(2*128), blkG, 0, stream>>>(xcat, 1536, WcombT, 1536, gbias,
                                                       gates, 4*HH, (_Float16*)nullptr, 0, 2, 128, 1536, 0);
        // LSTM cell
        k_cell<<<grid1((size_t)BB*HH), blk, 0, stream>>>(gates, hOut, cOut, h16);
        // logits_t = h_new @ fcW + fcb  (32x512x10000), written into d_out[b][t][:]
        k_gemm_wmma<<<gridT(2*625), blkG, 0, stream>>>(h16, HH, fcWT, HH, fcb,
                                                       out + (size_t)t * VV, TT*VV,
                                                       (_Float16*)nullptr, 0, 2, 625, HH, 0);
    }
}
